// decoder_plus_71047349010995
// MI455X (gfx1250) — matrix operand order repaired and checked by reading, not compiled, not run on hardware
//
#include <hip/hip_runtime.h>

// ---------------------------------------------------------------------------
// Types for CDNA5 WMMA (wave32, 16x16x32 bf16 -> f32) and TDM descriptors
// ---------------------------------------------------------------------------
typedef unsigned short u16t;
typedef __attribute__((ext_vector_type(16))) __bf16 v16bf;
typedef __attribute__((ext_vector_type(16))) u16t   v16u;
typedef __attribute__((ext_vector_type(8)))  float  v8f;
typedef __attribute__((ext_vector_type(4)))  unsigned int u32x4;
typedef __attribute__((ext_vector_type(8)))  int    i32x8;
typedef __attribute__((ext_vector_type(4)))  int    i32x4;

static __device__ __forceinline__ u16t f2bf(float f) {
  union { float f; unsigned u; } cv; cv.f = f;
  unsigned u = cv.u;
  return (u16t)((u + 0x7fffu + ((u >> 16) & 1u)) >> 16);  // RNE f32->bf16
}

static __device__ __forceinline__ v8f wmma_bf16(v16u a, v16u b, v8f c) {
  return __builtin_amdgcn_wmma_f32_16x16x32_bf16(
      false, __builtin_bit_cast(v16bf, a),
      false, __builtin_bit_cast(v16bf, b),
      (short)0, c, false, false);
}

// ---------------------------------------------------------------------------
// TDM: DMA a 16-row x K-col f32 weight tile (row-major, row stride K) into LDS.
// tensor_dim1 = rows remaining in W from this tile -> OOB rows are hardware-
// zero-filled, giving free padding for N not a multiple of 16.
// D# layout per CDNA5 ISA ch.8 (group0: count/lds_addr/global_addr/type=2;
// group1: data_size=4B, dims/strides; groups 2-3 zero for a 2D tile).
// ---------------------------------------------------------------------------
#if __has_builtin(__builtin_amdgcn_tensor_load_to_lds)
#define USE_TDM 1
static __device__ __forceinline__ void tdm_load_w_tile(
    const float* gsrc, float* ldsDst, int K, int nrem)
{
  unsigned long long ga = (unsigned long long)gsrc;
  unsigned ldsOff = (unsigned)(unsigned long long)(void*)ldsDst;
  u32x4 g0 = { 1u,                                   // count=1, user desc
               ldsOff,                               // lds_addr
               (unsigned)ga,                         // global_addr[31:0]
               (unsigned)((ga >> 32) & 0x1FFFFFFu) | 0x80000000u }; // [56:32] | type=2
  i32x8 g1 = { 0x00020000,                 // data_size=2 (4B), mask=0, no pad/iter
               (int)((K & 0xFFFF) << 16),  // tensor_dim0 = K  (bits 79:48)
               (int)((nrem & 0xFFFF) << 16), // tensor_dim1 = nrem (bits 111:80)
               (int)(K << 16),             // tile_dim0 = K    (bits 127:112)
               16,                         // tile_dim1 = 16, tile_dim2 = 0
               K,                          // tensor_dim0_stride = K (bits 207:160)
               0, 0 };                     // dim1_stride = 0 (2D tile)
  i32x4 gz = { 0, 0, 0, 0 };
#if defined(__clang_major__) && __clang_major__ >= 23
  i32x8 gz8 = { 0, 0, 0, 0, 0, 0, 0, 0 };
  __builtin_amdgcn_tensor_load_to_lds(g0, g1, gz, gz, gz8, 0);
#else
  __builtin_amdgcn_tensor_load_to_lds(g0, g1, gz, gz, 0);
#endif
  __builtin_amdgcn_s_wait_tensorcnt(0);
}
#else
#define USE_TDM 0
#endif

// ---------------------------------------------------------------------------
// Generic 1x1-conv GEMM:  out[b,n,p] = PReLU( sum_k A[b,k,p] * W[n,k] + bias[n] )
// A: (B,K,P) NCHW f32, W: (N,K) f32.  One wave = one 16(M)x16(N) tile.
// A fragment per ISA table: lane m=L%16, half=L/16, elem e -> K = e+8*half+8*(e>=8)
// B fragment: lane n=L%16, half=L/16, elem e -> K = e+8*half+8*(e>=8)
// Weight tile staged via TDM (TENSORcnt), re-packed to bf16 fragment layout.
// ---------------------------------------------------------------------------
__global__ __launch_bounds__(128) void conv1x1_wmma(
    const float* __restrict__ A, const float* __restrict__ W,
    const float* __restrict__ bias, const float* __restrict__ alpha,
    float* __restrict__ out, int K, int N, int P)
{
  __shared__ float ldsWf[96 * 16];         // TDM landing: [n][k] f32 (max K 96)
  __shared__ u16t  ldsW[96 * 16];          // fragment layout: [k][n] bf16
  const int tid  = threadIdx.x;
  const int lane = tid & 31;
  const int wave = tid >> 5;
  const int ntile = blockIdx.y;

#if USE_TDM
  if (wave == 0)
    tdm_load_w_tile(W + (long)ntile * 16 * K, ldsWf, K, N - ntile * 16);
  __syncthreads();
  for (int i = tid; i < K * 16; i += 128) {      // repack [n][k]f32 -> [k][n]bf16
    int n = i / K, k = i - n * K;
    ldsW[k * 16 + n] = f2bf(ldsWf[i]);
  }
#else
  for (int i = tid; i < K * 16; i += 128) {
    int k = i >> 4, n = i & 15;
    int ch = ntile * 16 + n;
    ldsW[i] = f2bf((ch < N) ? W[ch * K + k] : 0.f);
  }
#endif
  __syncthreads();

  const int mtile = blockIdx.x * 4 + wave;
  const int mrow  = mtile * 16 + (lane & 15);
  const int half  = lane >> 4;
  const int b = mrow / P;
  const int p = mrow - b * P;
  const float* Ab = A + (long)b * K * P + p;

  v8f acc = {};
  for (int kk = 0; kk < K; kk += 32) {
    if (kk + 32 < K) __builtin_prefetch(Ab + (long)(kk + 32) * P, 0, 1);
    v16u a, bf;
#pragma unroll
    for (int e = 0; e < 16; ++e) {
      int kA = kk + e + 8 * half + ((e >= 8) ? 8 : 0);
      a[e]  = f2bf(Ab[(long)kA * P]);
      int kB = kk + e + 8 * half + ((e >= 8) ? 8 : 0);
      bf[e] = ldsW[kB * 16 + (lane & 15)];
    }
    acc = wmma_bf16(a, bf, acc);
  }

  const int n  = lane & 15;
  const int ch = ntile * 16 + n;
  if (ch < N) {
    float bv = bias ? bias[ch] : 0.f;
    float al = alpha ? alpha[0] : 1.f;
#pragma unroll
    for (int j = 0; j < 8; ++j) {
      int m  = mtile * 16 + j + 8 * half;
      int bb = m / P, pp = m - bb * P;
      float x = acc[j] + bv;
      if (alpha) x = (x > 0.f) ? x : al * x;
      out[((long)bb * N + ch) * P + pp] = x;
    }
  }
}

// ---------------------------------------------------------------------------
// Build 96-channel fusion input at 32x32: [fea_high, down2(fea_mid), down4(fea_low)]
// bilinear antialias=False downsample => fixed 2x2 averages.
// ---------------------------------------------------------------------------
__global__ __launch_bounds__(256) void prep_x96(
    const float* __restrict__ fh, const float* __restrict__ fm,
    const float* __restrict__ fl, float* __restrict__ x96)
{
  int i = blockIdx.x * 256 + threadIdx.x;
  if (i >= 8 * 96 * 1024) return;
  int p = i & 1023, c = (i >> 10) % 96, b = i / (96 * 1024);
  int y = p >> 5, x = p & 31;
  float v;
  if (c < 32) {
    v = fh[(((long)b * 32 + c) * 32 + y) * 32 + x];
  } else if (c < 64) {
    const float* s = fm + ((long)b * 32 + (c - 32)) * 4096;
    int yy = 2 * y, xx = 2 * x;
    v = 0.25f * (s[yy * 64 + xx] + s[yy * 64 + xx + 1] +
                 s[(yy + 1) * 64 + xx] + s[(yy + 1) * 64 + xx + 1]);
  } else {
    const float* s = fl + ((long)b * 32 + (c - 64)) * 16384;
    int yy = 4 * y + 1, xx = 4 * x + 1;   // src coord 4i+1.5 -> pixels 4i+1,4i+2
    v = 0.25f * (s[yy * 128 + xx] + s[yy * 128 + xx + 1] +
                 s[(yy + 1) * 128 + xx] + s[(yy + 1) * 128 + xx + 1]);
  }
  x96[i] = v;
}

// Global average pool of fused (8,32,32,32) -> gapv (8,32)
__global__ __launch_bounds__(256) void gap_kernel(
    const float* __restrict__ fused, float* __restrict__ gapv)
{
  __shared__ float red[256];
  const float* s = fused + (long)blockIdx.x * 1024;
  float sum = 0.f;
  for (int i = threadIdx.x; i < 1024; i += 256) sum += s[i];
  red[threadIdx.x] = sum; __syncthreads();
  for (int st = 128; st > 0; st >>= 1) {
    if (threadIdx.x < st) red[threadIdx.x] += red[threadIdx.x + st];
    __syncthreads();
  }
  if (threadIdx.x == 0) gapv[blockIdx.x] = red[0] * (1.f / 1024.f);
}

// kl[b,k,y,x] = softmax_k( pixel_shuffle_r(conv)[b,k,y,x] + add[b,k,y,x] )
__global__ __launch_bounds__(256) void softmax9(
    const float* __restrict__ convbuf, int Cc, int r,
    const float* __restrict__ addbuf, float* __restrict__ out, int H, int total)
{
  int i = blockIdx.x * 256 + threadIdx.x;
  if (i >= total) return;
  int x = i % H, y = (i / H) % H, b = i / (H * H);
  float v[9], mx = -1e30f;
#pragma unroll
  for (int k = 0; k < 9; ++k) {
    int c = k * r * r + (y % r) * r + (x % r);
    float val = convbuf[(((long)b * Cc + c) * 32 + y / r) * 32 + x / r];
    if (addbuf) val += addbuf[(((long)b * 9 + k) * H + y) * H + x];
    v[k] = val; mx = fmaxf(mx, val);
  }
  float s = 0.f;
#pragma unroll
  for (int k = 0; k < 9; ++k) { v[k] = __expf(v[k] - mx); s += v[k]; }
  float inv = 1.f / s;
#pragma unroll
  for (int k = 0; k < 9; ++k)
    out[(((long)b * 9 + k) * H + y) * H + x] = v[k] * inv;
}

// AdaptiveAvgPool2d(3) of zero-padded (H+2) image of
//   val(y,x) = pixel_shuffle_r(src)[b,c,y,x] + (base ? base[b,c,y,x] : 0)
// (pool/conv1x1 order swapped vs reference — exact since conv1x1 is linear)
__global__ __launch_bounds__(64) void pool3(
    const float* __restrict__ src, int r, const float* __restrict__ base,
    int H, float* __restrict__ pooled)
{
  int blk = blockIdx.x;
  int bin = blk % 9, c = (blk / 9) % 32, b = blk / (9 * 32);
  int Hp = H + 2;
  int bh = bin / 3, bw = bin % 3;
  int h0 = bh * Hp / 3, h1 = ((bh + 1) * Hp + 2) / 3;
  int w0 = bw * Hp / 3, w1 = ((bw + 1) * Hp + 2) / 3;
  int nh = h1 - h0, nw = w1 - w0, tot = nh * nw;
  float sum = 0.f;
  for (int t = threadIdx.x; t < tot; t += 64) {
    int y = h0 + t / nw - 1, x = w0 + t % nw - 1;   // padded -> orig coords
    if (y >= 0 && y < H && x >= 0 && x < H) {
      int c2 = c * r * r + (y % r) * r + (x % r);
      float v = src[(((long)b * (32 * r * r) + c2) * 32 + y / r) * 32 + x / r];
      if (base) v += base[(((long)b * 32 + c) * H + y) * H + x];
      sum += v;
    }
  }
  __shared__ float red[64];
  red[threadIdx.x] = sum; __syncthreads();
  for (int st = 32; st > 0; st >>= 1) {
    if (threadIdx.x < st) red[threadIdx.x] += red[threadIdx.x + st];
    __syncthreads();
  }
  if (threadIdx.x == 0) pooled[blk] = red[0] / (float)(nh * nw);
}

// Two tiny 32->32 matmuls on the 3x3 pooled grid: kc = PReLU(W1*(W0*pooled+b0)+b1)
__global__ __launch_bounds__(288) void gen_small(
    const float* __restrict__ pooled, const float* __restrict__ W0,
    const float* __restrict__ b0, const float* __restrict__ W1,
    const float* __restrict__ b1, const float* __restrict__ al,
    float* __restrict__ kc)
{
  __shared__ float pl[288], yb[288];
  int b = blockIdx.x, t = threadIdx.x;
  pl[t] = pooled[b * 288 + t];
  __syncthreads();
  int co = t / 9, pos = t % 9;
  float acc = b0[co];
  for (int c = 0; c < 32; ++c) acc += W0[co * 32 + c] * pl[c * 9 + pos];
  yb[t] = acc;
  __syncthreads();
  acc = b1[co];
  for (int c = 0; c < 32; ++c) acc += W1[co * 32 + c] * yb[c * 9 + pos];
  float a = al[0];
  kc[b * 288 + t] = (acc > 0.f) ? acc : a * acc;
}

// CARAFE-style reassembly: both spatial-kernel and per-channel-kernel outputs.
__global__ __launch_bounds__(256) void reassemble(
    const float* __restrict__ fea, const float* __restrict__ ks,
    const float* __restrict__ kc, const float* __restrict__ gapv,
    float* __restrict__ out_sp, float* __restrict__ out_ch, int H, int total)
{
  int i = blockIdx.x * 256 + threadIdx.x;
  if (i >= total) return;
  int x = i % H, y = (i / H) % H, c = (i / (H * H)) % 32, b = i / (32 * H * H);
  float g = gapv[b * 32 + c];
  const float* F = fea + ((long)b * 32 + c) * H * H;
  float asp = 0.f, ach = 0.f;
#pragma unroll
  for (int k = 0; k < 9; ++k) {
    int yy = y + k / 3 - 1, xx = x + k % 3 - 1;
    float v = (yy >= 0 && yy < H && xx >= 0 && xx < H) ? (F[yy * H + xx] + g) : 0.f;
    asp += v * ks[(((long)b * 9 + k) * H + y) * H + x];
    ach += v * kc[((long)b * 32 + c) * 9 + k];
  }
  out_sp[i] = asp; out_ch[i] = ach;
}

// ---------------------------------------------------------------------------
// Final conv 192->32 @128x128 with the bilinear-upsample of the six 32-channel
// sources FUSED into the WMMA A-fragment build (no 100MB concat buffer; all
// gathers hit the L2-resident source tensors). Weights staged via TDM.
// ---------------------------------------------------------------------------
static __device__ __forceinline__ void prep_s(float s, int n, int& i0, int& i1, float& f) {
  int t = (int)floorf(s);
  f = s - (float)t;
  i0 = min(max(t, 0), n - 1);
  i1 = min(max(t + 1, 0), n - 1);
}
static __device__ __forceinline__ float lerp2(const float* P, int n,
    int y0, int y1, float fy, int x0, int x1, float fx) {
  float a = P[y0 * n + x0], b = P[y0 * n + x1];
  float c = P[y1 * n + x0], d = P[y1 * n + x1];
  return (a * (1.f - fx) + b * fx) * (1.f - fy) + (c * (1.f - fx) + d * fx) * fy;
}

__global__ __launch_bounds__(128) void final_conv_wmma(
    const float* __restrict__ hs, const float* __restrict__ hc,
    const float* __restrict__ mn, const float* __restrict__ mc,
    const float* __restrict__ ln, const float* __restrict__ lc,
    const float* __restrict__ Wf, const float* __restrict__ af,
    float* __restrict__ out)
{
  __shared__ float ldsWf[192 * 16];   // TDM landing [n][k] f32
  __shared__ u16t  ldsW[192 * 16];    // fragment layout [k][n] bf16
  const int tid = threadIdx.x, lane = tid & 31, wave = tid >> 5;
  const int ntile = blockIdx.y;

#if USE_TDM
  if (wave == 0)
    tdm_load_w_tile(Wf + (long)ntile * 16 * 192, ldsWf, 192, 32 - ntile * 16);
  __syncthreads();
  for (int i = tid; i < 192 * 16; i += 128) {
    int n = i / 192, k = i - n * 192;
    ldsW[k * 16 + n] = f2bf(ldsWf[i]);
  }
#else
  for (int i = tid; i < 192 * 16; i += 128) {
    int k = i >> 4, n = i & 15;
    ldsW[i] = f2bf(Wf[(ntile * 16 + n) * 192 + k]);
  }
#endif
  __syncthreads();

  const int mtile = blockIdx.x * 4 + wave;
  const int mrow  = mtile * 16 + (lane & 15);
  const int half  = lane >> 4;
  const int b = mrow >> 14;
  const int p = mrow & 16383;
  const int y = p >> 7, x = p & 127;

  int y40, y41, x40, x41, y20, y21, x20, x21;
  float fy4, fx4, fy2, fx2;
  prep_s(0.25f * y - 0.375f, 32, y40, y41, fy4);   // up x4: 32 -> 128
  prep_s(0.25f * x - 0.375f, 32, x40, x41, fx4);
  prep_s(0.50f * y - 0.250f, 64, y20, y21, fy2);   // up x2: 64 -> 128
  prep_s(0.50f * x - 0.250f, 64, x20, x21, fx2);

  const float* srcs[6] = { hs, hc, mn, mc, ln, lc };
  v8f acc = {};
#pragma unroll
  for (int g = 0; g < 6; ++g) {            // K-step 32 == one 32-channel group
    const float* S = srcs[g];
    v16u a, bf;
#pragma unroll
    for (int e = 0; e < 16; ++e) {
      int c = e + 8 * half + ((e >= 8) ? 8 : 0);   // channel within group
      float v;
      if (g >= 4)       v = S[(((long)b * 32 + c) << 14) + p];
      else if (g >= 2)  v = lerp2(S + (((long)b * 32 + c) << 12), 64,
                                  y20, y21, fy2, x20, x21, fx2);
      else              v = lerp2(S + (((long)b * 32 + c) << 10), 32,
                                  y40, y41, fy4, x40, x41, fx4);
      a[e]  = f2bf(v);
      int kB = g * 32 + e + 8 * half + ((e >= 8) ? 8 : 0);
      bf[e] = ldsW[kB * 16 + (lane & 15)];
    }
    acc = wmma_bf16(a, bf, acc);
  }

  const int ch = ntile * 16 + (lane & 15);
  const float al = af[0];
#pragma unroll
  for (int j = 0; j < 8; ++j) {
    int m = mtile * 16 + j + 8 * half;
    int bb = m >> 14, pp = m & 16383;
    float xo = acc[j];
    xo = (xo > 0.f) ? xo : al * xo;
    out[(((long)bb * 32 + ch) << 14) + pp] = xo;
  }
}

// ---------------------------------------------------------------------------
// Host orchestration
// ---------------------------------------------------------------------------
extern "C" void kernel_launch(void* const* d_in, const int* in_sizes, int n_in,
                              void* d_out, int out_size, void* d_ws, size_t ws_size,
                              hipStream_t stream) {
  const float* fea_low  = (const float*)d_in[0];
  const float* fea_mid  = (const float*)d_in[1];
  const float* fea_high = (const float*)d_in[2];
  const float *Wfh=(const float*)d_in[3],  *bfh=(const float*)d_in[4],  *afh=(const float*)d_in[5];
  const float *Wcl=(const float*)d_in[6],  *bcl=(const float*)d_in[7],  *acl=(const float*)d_in[8];
  const float *Wcm=(const float*)d_in[9],  *bcm=(const float*)d_in[10], *acm=(const float*)d_in[11];
  const float *Wch=(const float*)d_in[12], *bch=(const float*)d_in[13], *ach=(const float*)d_in[14];
  const float *Wls=(const float*)d_in[15], *bls=(const float*)d_in[16], *als=(const float*)d_in[17];
  const float *Wms=(const float*)d_in[18], *bms=(const float*)d_in[19], *ams=(const float*)d_in[20];
  const float *Wrl=(const float*)d_in[21], *brl=(const float*)d_in[22], *arl=(const float*)d_in[23];
  const float *Wrm=(const float*)d_in[24], *brm=(const float*)d_in[25], *arm=(const float*)d_in[26];
  const float *Wg0l=(const float*)d_in[27],*bg0l=(const float*)d_in[28],*Wg1l=(const float*)d_in[29],*bg1l=(const float*)d_in[30],*agl=(const float*)d_in[31];
  const float *Wg0m=(const float*)d_in[32],*bg0m=(const float*)d_in[33],*Wg1m=(const float*)d_in[34],*bg1m=(const float*)d_in[35],*agm=(const float*)d_in[36];
  const float *Wg0h=(const float*)d_in[37],*bg0h=(const float*)d_in[38],*Wg1h=(const float*)d_in[39],*bg1h=(const float*)d_in[40],*agh=(const float*)d_in[41];
  const float *Wf=(const float*)d_in[42],  *af=(const float*)d_in[43];

  float* w = (float*)d_ws;
  size_t off = 0;
  auto alloc = [&](size_t n) { float* p = w + off; off += n; return p; };
  float* x96      = alloc(786432);    // (8,96,32,32)
  float* fused    = alloc(262144);    // (8,32,32,32)
  float* gapv     = alloc(256);       // (8,32)
  float* cl       = alloc(1179648);   // (8,144,32,32)
  float* cm       = alloc(294912);    // (8,36,32,32)
  float* chs      = alloc(73728);     // (8,9,32,32)
  float* rl       = alloc(4194304);   // (8,512,32,32)
  float* rm       = alloc(1048576);   // (8,128,32,32)
  float* lsb      = alloc(1179648);   // (8,9,128,128)
  float* msb      = alloc(294912);    // (8,9,64,64)
  float* kl       = alloc(1179648);
  float* km       = alloc(294912);
  float* kh       = alloc(73728);
  float* pooled_l = alloc(2304);      // (8,32,9)
  float* pooled_m = alloc(2304);
  float* pooled_h = alloc(2304);
  float* kcl      = alloc(2304);
  float* kcm      = alloc(2304);
  float* kch      = alloc(2304);
  float* low_new  = alloc(4194304);   // (8,32,128,128)
  float* ch_low   = alloc(4194304);
  float* mid_new  = alloc(1048576);   // (8,32,64,64)
  float* ch_mid   = alloc(1048576);
  float* high_new = alloc(262144);    // (8,32,32,32)
  float* ch_high  = alloc(262144);

  // 1) fusion input + fused = PReLU(conv 96->32) @ 32x32
  prep_x96<<<(786432 + 255) / 256, 256, 0, stream>>>(fea_high, fea_mid, fea_low, x96);
  conv1x1_wmma<<<dim3(128, 2), 128, 0, stream>>>(x96, Wfh, bfh, afh, fused, 96, 32, 1024);

  // 2) GAP
  gap_kernel<<<256, 256, 0, stream>>>(fused, gapv);

  // 3) WMMA 1x1 convs from fused (M=8192) and from low/mid features
  conv1x1_wmma<<<dim3(128, 9), 128, 0, stream>>>(fused, Wcl, bcl, acl, cl, 32, 144, 1024);
  conv1x1_wmma<<<dim3(128, 3), 128, 0, stream>>>(fused, Wcm, bcm, acm, cm, 32, 36, 1024);
  conv1x1_wmma<<<dim3(128, 1), 128, 0, stream>>>(fused, Wch, bch, ach, chs, 32, 9, 1024);
  conv1x1_wmma<<<dim3(128, 32), 128, 0, stream>>>(fused, Wrl, brl, arl, rl, 32, 512, 1024);
  conv1x1_wmma<<<dim3(128, 8), 128, 0, stream>>>(fused, Wrm, brm, arm, rm, 32, 128, 1024);
  conv1x1_wmma<<<dim3(2048, 1), 128, 0, stream>>>(fea_low, Wls, bls, als, lsb, 32, 9, 16384);
  conv1x1_wmma<<<dim3(512, 1), 128, 0, stream>>>(fea_mid, Wms, bms, ams, msb, 32, 9, 4096);

  // 4) softmax-normalized spatial kernels (pixel-shuffle gathered in-kernel)
  softmax9<<<(131072 + 255) / 256, 256, 0, stream>>>(cl, 144, 4, lsb, kl, 128, 131072);
  softmax9<<<(32768 + 255) / 256, 256, 0, stream>>>(cm, 36, 2, msb, km, 64, 32768);
  softmax9<<<(8192 + 255) / 256, 256, 0, stream>>>(chs, 9, 1, nullptr, kh, 32, 8192);

  // 5) adaptive-pool-3 of padded (pixel_shuffle(r)+base), then tiny dual matmul
  pool3<<<2304, 64, 0, stream>>>(rl, 4, fea_low, 128, pooled_l);
  pool3<<<2304, 64, 0, stream>>>(rm, 2, fea_mid, 64, pooled_m);
  pool3<<<2304, 64, 0, stream>>>(fused, 1, nullptr, 32, pooled_h);
  gen_small<<<8, 288, 0, stream>>>(pooled_l, Wg0l, bg0l, Wg1l, bg1l, agl, kcl);
  gen_small<<<8, 288, 0, stream>>>(pooled_m, Wg0m, bg0m, Wg1m, bg1m, agm, kcm);
  gen_small<<<8, 288, 0, stream>>>(pooled_h, Wg0h, bg0h, Wg1h, bg1h, agh, kch);

  // 6) reassembly (spatial + per-channel dynamic kernels)
  reassemble<<<16384, 256, 0, stream>>>(fea_low, kl, kcl, gapv, low_new, ch_low, 128, 4194304);
  reassemble<<<4096, 256, 0, stream>>>(fea_mid, km, kcm, gapv, mid_new, ch_mid, 64, 1048576);
  reassemble<<<1024, 256, 0, stream>>>(fea_high, kh, kch, gapv, high_new, ch_high, 32, 262144);

  // 7) fused upsample + concat + conv 192->32 + PReLU straight into d_out
  final_conv_wmma<<<dim3(2048, 2), 128, 0, stream>>>(
      high_new, ch_high, mid_new, ch_mid, low_new, ch_low, Wf, af, (float*)d_out);
}
